// SLCPTN_77816217469306
// MI455X (gfx1250) — matrix operand order repaired and checked by reading, not compiled, not run on hardware
//
#include <hip/hip_runtime.h>
#include <hip/hip_bf16.h>

// ---------------------------------------------------------------------------
// CDNA5 (gfx1250): bf16 WMMA (v_wmma_f32_16x16x32_bf16) for all GEMM-shaped
// work; convs are implicit GEMMs over halo-padded channel-last activations.
// A-tiles are staged with GLOBAL_LOAD_ASYNC_TO_LDS_B128 (ASYNCcnt), double-
// buffered so copies overlap WMMA; BN folded into weights; f32 VALU epilogues.
// ---------------------------------------------------------------------------

#if defined(__has_builtin)
#if __has_builtin(__builtin_amdgcn_global_load_async_to_lds_b128) && \
    __has_builtin(__builtin_amdgcn_s_wait_asynccnt)
#define HAVE_ASYNC_LDS 1
#endif
#endif
#ifndef HAVE_ASYNC_LDS
#define HAVE_ASYNC_LDS 0
#endif

typedef __attribute__((ext_vector_type(16))) __bf16 bf16x16;
typedef __attribute__((ext_vector_type(8)))  __bf16 bf16x8;
typedef __attribute__((ext_vector_type(8)))  float  f32x8;

// async-copy pointer types (builtin signature: v4i addrspace(1)* src,
// v4i addrspace(3)* LDS dst, imm offset, imm cpol)
typedef int v4i __attribute__((vector_size(4 * sizeof(int))));
typedef __attribute__((address_space(1))) v4i gv4i;
typedef __attribute__((address_space(3))) v4i lv4i;

__device__ __forceinline__ f32x8 zero8() {
  f32x8 z;
#pragma unroll
  for (int j = 0; j < 8; ++j) z[j] = 0.f;
  return z;
}

__device__ __forceinline__ f32x8 wmma_bf16(bf16x16 a, bf16x16 b, f32x8 c) {
  return __builtin_amdgcn_wmma_f32_16x16x32_bf16(false, a, false, b, (short)0, c,
                                                 false, false);
}

// A-fragment (16x32 bf16, wave32): lane row = lane&15, half = lane>>4.
// Element j -> K = half*16 + j: one contiguous 16-element run per lane,
// loaded as two 8-element (16B) pieces.
__device__ __forceinline__ bf16x16 frag_a(const __bf16* row, int half) {
  bf16x8 lo = *(const bf16x8*)(row + half * 16);
  bf16x8 hi = *(const bf16x8*)(row + half * 16 + 8);
  bf16x16 r;
#pragma unroll
  for (int j = 0; j < 8; ++j) { r[j] = lo[j]; r[j + 8] = hi[j]; }
  return r;
}

// ---------------------------------------------------------------------------
// Generic implicit-GEMM conv kernel.
//   Out[m, co] = act( sum_k A[m, k] * W[k, co] + bias[co] )
//   m = b*Lout + l ; k = t*CIN + c.
// Scalar mode (ASYNCA=false): A[m,k] = X[b, l*strideL - pad + t, c], zero
//   outside [0,Lin); Lin/pad natural meaning.
// Async mode (ASYNCA=true): X halo-padded (reads in-bounds, 16B aligned);
//   `Lin` = PADDED per-batch rows, `pad` = HALO - real_pad. A-tiles staged
//   global->LDS with the gfx1250 async-copy path, DOUBLE-BUFFERED so the
//   next tile's copy overlaps this tile's WMMAs. K-overhang rows of A may be
//   garbage but the B tile is zero there, so products vanish.
// Blocks never straddle batches (Lout % 128 == 0 for KTAPS>1), so the batch
// decomposition is hoisted to one division per thread.
// Output rows remapped b*LoutP + l. Block tile 128(M) x 64(N): 8 waves as
// 4x2, each computing a 2x2 grid of 16x16 WMMA tiles. K step 32.
// ACT: 0 = none, 1 = relu, 2 = exact gelu.
// ---------------------------------------------------------------------------
template <int CIN, int KTAPS, typename AT, typename OT, int ACT, bool ASYNCA>
__global__ __launch_bounds__(256) void conv_gemm(
    const AT* __restrict__ X, const __bf16* __restrict__ W,
    const float* __restrict__ bias, OT* __restrict__ Out, int Lout, int LoutP,
    int Lin, int strideL, int pad, int CinStride, int Kreal, int Nreal,
    int OutStride, int OutOffset) {
  __shared__ __bf16 At[2][128][32];
  __shared__ __bf16 Bt[2][64][32];  // transposed: Bt[buf][n][k]
  const int tid = threadIdx.x;
  const int wave = tid >> 5, lane = tid & 31;
  const int half = lane >> 4, l16 = lane & 15;
  const int wm = wave >> 1, wn = wave & 1;
  const int m0 = blockIdx.y * 128, n0 = blockIdx.x * 64;
  // hoisted batch decomposition (one division per thread)
  int bb0 = 0, l0 = 0;
  if (KTAPS > 1) { bb0 = m0 / Lout; l0 = m0 - bb0 * Lout; }
  f32x8 acc[2][2];
  acc[0][0] = zero8(); acc[0][1] = zero8();
  acc[1][0] = zero8(); acc[1][1] = zero8();
  const int KP = (Kreal + 31) & ~31;
  const int T = KP >> 5;

  auto stageA = [&](int t, int buf) {
    const int k0 = t * 32;
    if (ASYNCA) {
      // 128x32 bf16 tile = 512 x 16B chunks, async global->LDS
      for (int ch = tid; ch < 512; ch += 256) {
        int i = ch >> 2, j0 = (ch & 3) << 3;
        int kk = k0 + j0;
        size_t srcoff;
        if (KTAPS == 1) {
          srcoff = (size_t)(m0 + i) * (size_t)CinStride + kk;
        } else {
          int tt = kk / CIN, c = kk - tt * CIN;
          int pos = (l0 + i) * strideL + tt + pad;  // pad == HALO - real_pad
          srcoff = ((size_t)bb0 * Lin + pos) * (size_t)CinStride + c;
        }
#if HAVE_ASYNC_LDS
        __builtin_amdgcn_global_load_async_to_lds_b128(
            (gv4i*)(X + srcoff), (lv4i*)&At[buf][i][j0], 0, 0);
#else
#pragma unroll
        for (int q = 0; q < 8; ++q)
          At[buf][i][j0 + q] = (__bf16)(float)X[srcoff + q];
#endif
      }
    } else {
      // scalar staging with im2col bounds + bf16 convert
      for (int e = tid; e < 128 * 32; e += 256) {
        int i = e >> 5, j = e & 31;
        int kk = k0 + j;
        float v = 0.f;
        if (kk < Kreal) {
          if (KTAPS == 1) {
            v = (float)X[(size_t)(m0 + i) * (size_t)CinStride + kk];
          } else {
            int tt = kk / CIN, c = kk - tt * CIN;
            int pos = (l0 + i) * strideL - pad + tt;
            if (pos >= 0 && pos < Lin)
              v = (float)X[((size_t)bb0 * Lin + pos) * (size_t)CinStride + c];
          }
        }
        At[buf][i][j] = (__bf16)v;
      }
    }
  };
  auto stageB = [&](int t, int buf) {
    const int k0 = t * 32;
    for (int e = tid; e < 64 * 32; e += 256) {
      int n = e >> 5, j = e & 31;
      int kk = k0 + j, gn = n0 + n;
      Bt[buf][n][j] = (kk < Kreal && gn < Nreal) ? W[(size_t)kk * Nreal + gn]
                                                 : (__bf16)0.f;
    }
  };

  // prologue: stage tile 0 into buffer 0
  stageA(0, 0);
  stageB(0, 0);
  for (int t = 0; t < T; ++t) {
    const int cur = t & 1, nxt = cur ^ 1;
    __syncthreads();  // iter t-1 consumers of buffer `nxt` are done
    if (t + 1 < T) {  // overlap: stage tile t+1 while computing tile t
      stageA(t + 1, nxt);
      stageB(t + 1, nxt);
    }
#if HAVE_ASYNC_LDS
    if (ASYNCA) {
      if (t + 1 < T) __builtin_amdgcn_s_wait_asynccnt(2);  // tile t landed
      else           __builtin_amdgcn_s_wait_asynccnt(0);
    }
#endif
    __syncthreads();  // tile t visible to all waves
    bf16x16 af0 = frag_a(&At[cur][wm * 32 + l16][0], half);
    bf16x16 af1 = frag_a(&At[cur][wm * 32 + 16 + l16][0], half);
    bf16x16 bf0 = *(const bf16x16*)(&Bt[cur][wn * 32 + l16][half * 16]);
    bf16x16 bf1 = *(const bf16x16*)(&Bt[cur][wn * 32 + 16 + l16][half * 16]);
    acc[0][0] = wmma_bf16(af0, bf0, acc[0][0]);
    acc[0][1] = wmma_bf16(af0, bf1, acc[0][1]);
    acc[1][0] = wmma_bf16(af1, bf0, acc[1][0]);
    acc[1][1] = wmma_bf16(af1, bf1, acc[1][1]);
  }
  // epilogue: C fragment mapping M = half*8 + r, N = l16
  const size_t orow0 =
      (KTAPS == 1) ? (size_t)m0 : ((size_t)bb0 * LoutP + l0);
#pragma unroll
  for (int sn = 0; sn < 2; ++sn) {
    const int ncol = n0 + wn * 32 + sn * 16 + l16;
    if (ncol >= Nreal) continue;
    const float bv = bias[ncol];
#pragma unroll
    for (int sm = 0; sm < 2; ++sm) {
#pragma unroll
      for (int r = 0; r < 8; ++r) {
        float v = acc[sm][sn][r] + bv;
        if (ACT == 1) v = fmaxf(v, 0.f);
        if (ACT == 2) v = 0.5f * v * (1.f + erff(v * 0.70710678118654752f));
        int roff = wm * 32 + sm * 16 + half * 8 + r;
        Out[(orow0 + roff) * OutStride + OutOffset + ncol] = (OT)v;
      }
    }
  }
}

// ---------------------------------------------------------------------------
// Simplified-transformer layer: one block per sample (B=512). bf16 tokens.
// Tokens: 128 x 192, H=4 heads, 4 sequential token-groups of 32.
// LDS overlays keep static shared memory <= 64KB:
//   smA: qkb (32x384 bf16); after QK it is dead and is overlaid with
//        attnb (4x32x32 bf16) at offset 0 and featT (192x32 bf16) at +8192.
//   smB: Bt (384x32 bf16) overlaid with scores (4x32x32 f32, used after Bt)
// ---------------------------------------------------------------------------
__global__ __launch_bounds__(256) void transformer_layer(
    const __bf16* __restrict__ in, const __bf16* __restrict__ wqk,
    const float* __restrict__ bqk, __bf16* __restrict__ out) {
  __shared__ __bf16 feat[32 * 192];
  __shared__ __align__(32) char smA[32 * 384 * 2];
  __shared__ __align__(32) char smB[384 * 32 * 2];
  __bf16* qkb = (__bf16*)smA;
  __bf16* attnb = (__bf16*)smA;            // overlay (qkb dead when written)
  __bf16* featT = (__bf16*)(smA + 8192);   // overlay: transposed V, 192x32
  __bf16* Bt = (__bf16*)smB;
  float* scores = (float*)smB;             // overlay (Bt dead when written)
  const int tid = threadIdx.x;
  const int wave = tid >> 5, lane = tid & 31;
  const int half = lane >> 4, l16 = lane & 15;
  const int b = blockIdx.x;
  const float qscale = 0.14433756729740643f;  // 48^-0.5

  {  // group 0 features
    const __bf16* inp = in + (size_t)b * 128 * 192;
    for (int e = tid; e < 32 * 192; e += 256) feat[e] = inp[e];
  }
  for (int i = 0; i < 4; ++i) {
    if (i > 0) {  // feat = prev attention output + next group input
      const __bf16* inp = in + ((size_t)b * 128 + i * 32) * 192;
      for (int e = tid; e < 32 * 192; e += 256)
        feat[e] = (__bf16)((float)feat[e] + (float)inp[e]);
    }
    __syncthreads();

    // ---- GEMM1: qk(32x384) = feat(32x192) @ wqk + bqk (q part scaled) ----
    f32x8 acc[6];
#pragma unroll
    for (int s = 0; s < 6; ++s) acc[s] = zero8();
    for (int k0 = 0; k0 < 192; k0 += 32) {
      __syncthreads();
      for (int e = tid; e < 384 * 32; e += 256) {
        int n = e >> 5, k = e & 31;
        Bt[n * 32 + k] = wqk[(size_t)(k0 + k) * 384 + n];
      }
      __syncthreads();
#pragma unroll
      for (int s = 0; s < 6; ++s) {
        int tI = s * 8 + wave, tm = tI & 1, tn = tI >> 1;
        bf16x16 af = frag_a(feat + (tm * 16 + l16) * 192 + k0, half);
        bf16x16 bfv = *(const bf16x16*)(Bt + (tn * 16 + l16) * 32 + half * 16);
        acc[s] = wmma_bf16(af, bfv, acc[s]);
      }
    }
#pragma unroll
    for (int s = 0; s < 6; ++s) {
      int tI = s * 8 + wave, tm = tI & 1, tn = tI >> 1;
#pragma unroll
      for (int r = 0; r < 8; ++r) {
        int m = tm * 16 + half * 8 + r, n = tn * 16 + l16;
        float v = acc[s][r] + bqk[n];
        if (n < 192) v *= qscale;  // q part pre-scaled
        qkb[m * 384 + n] = (__bf16)v;
      }
    }
    __syncthreads();

    // ---- QK^T: scores[h] = q_h(32x48) @ k_h^T, K padded 48->64.
    // Guards only remove whole 8/16-element runs, so use vector LDS loads. ---
#pragma unroll
    for (int u = 0; u < 2; ++u) {
      f32x8 a2 = zero8();
      int tI = wave + u * 8;
      int h = tI >> 2, tm = (tI >> 1) & 1, tn = tI & 1;
      const __bf16* arow = qkb + (tm * 16 + l16) * 384 + h * 48;
      const __bf16* brow = qkb + (tn * 16 + l16) * 384 + 192 + h * 48;
      {  // k0 = 0: all K in [0,32) < 48
        bf16x16 af = frag_a(arow, half);
        bf16x16 bfv = *(const bf16x16*)(brow + half * 16);
        a2 = wmma_bf16(af, bfv, a2);
      }
      {  // k0 = 32: A K=32+half*16+j; half 0 (K=32..47) valid, half 1 >= 48
        bf16x16 af;
        if (half == 0) {
          bf16x8 lo = *(const bf16x8*)(arow + 32);
          bf16x8 hi = *(const bf16x8*)(arow + 40);
#pragma unroll
          for (int j = 0; j < 8; ++j) { af[j] = lo[j]; af[j + 8] = hi[j]; }
        } else {
#pragma unroll
          for (int j = 0; j < 16; ++j) af[j] = (__bf16)0.f;
        }
        bf16x16 bfv;
        if (half == 0) {  // K = 32..47 valid
          bfv = *(const bf16x16*)(brow + 32);
        } else {          // K = 48..63 all out of range
#pragma unroll
          for (int j = 0; j < 16; ++j) bfv[j] = (__bf16)0.f;
        }
        a2 = wmma_bf16(af, bfv, a2);
      }
#pragma unroll
      for (int r = 0; r < 8; ++r) {
        int m = tm * 16 + half * 8 + r, n = tn * 16 + l16;
        scores[(h * 32 + m) * 32 + n] = a2[r];
      }
    }
    __syncthreads();

    // ---- softmax (threads 0..127) + build transposed V (threads 128..255) --
    if (tid < 128) {
      int h = tid >> 5, m = tid & 31;
      float* row = scores + (h * 32 + m) * 32;
      float mx = row[0];
      for (int j = 1; j < 32; ++j) mx = fmaxf(mx, row[j]);
      float sum = 0.f;
      for (int j = 0; j < 32; ++j) {
        float e = expf(row[j] - mx);
        row[j] = e;
        sum += e;
      }
      float inv = 1.f / sum;
      for (int j = 0; j < 32; ++j)
        attnb[(h * 32 + m) * 32 + j] = (__bf16)(row[j] * inv);
    } else {
      // featT[c*32 + k] = feat[k*192 + c]
      for (int e = tid - 128; e < 192 * 32; e += 128) {
        int c = e >> 5, k = e & 31;
        featT[e] = feat[k * 192 + c];
      }
    }
    __syncthreads();

    // ---- AV: out_h(32x48) = attn_h(32x32) @ V_h (V from featT, contiguous) --
    f32x8 acc3[3];
#pragma unroll
    for (int u = 0; u < 3; ++u) {
      int tI = wave + u * 8;
      int h = tI / 6, rem = tI - h * 6, tm = rem / 3, tn = rem - (rem / 3) * 3;
      bf16x16 af = frag_a(attnb + (h * 32 + tm * 16 + l16) * 32, half);
      int ncol = h * 48 + tn * 16 + l16;
      bf16x16 bfv = *(const bf16x16*)(featT + ncol * 32 + half * 16);
      acc3[u] = wmma_bf16(af, bfv, zero8());
    }
    __syncthreads();  // all V reads of feat/featT complete before overwrite
    __bf16* outp = out + ((size_t)b * 128 + i * 32) * 192;
#pragma unroll
    for (int u = 0; u < 3; ++u) {
      int tI = wave + u * 8;
      int h = tI / 6, rem = tI - h * 6, tm = rem / 3, tn = rem - (rem / 3) * 3;
#pragma unroll
      for (int r = 0; r < 8; ++r) {
        int m = tm * 16 + half * 8 + r, c = h * 48 + tn * 16 + l16;
        float v = acc3[u][r];
        feat[m * 192 + c] = (__bf16)v;     // carries to next group
        outp[(size_t)m * 192 + c] = (__bf16)v;  // group output
      }
    }
    __syncthreads();
  }
}

// ---------------------------------------------------------------------------
// Finalize: layernorm(a + h) per token, mean over 128 tokens, 192->3 matvec.
// 512 blocks x 192 threads (6 waves), wave32 shuffle reductions.
// ---------------------------------------------------------------------------
__global__ __launch_bounds__(192) void finalize_kernel(
    const __bf16* __restrict__ a, const float* __restrict__ h,
    const float* __restrict__ lng, const float* __restrict__ lnb,
    const float* __restrict__ wcls, const float* __restrict__ bcls,
    float* __restrict__ out) {
  const int b = blockIdx.x, c = threadIdx.x;
  const int wave = c >> 5, lane = c & 31;
  __shared__ float wsum[6], wsum2[6];
  __shared__ float accs[192];
  float acc = 0.f;
  const __bf16* ab = a + (size_t)b * 128 * 192;
  const float* hb = h + (size_t)b * 128 * 192;
  for (int t = 0; t < 128; ++t) {
    float v = (float)ab[t * 192 + c] + hb[t * 192 + c];
    float s = v, s2 = v * v;
#pragma unroll
    for (int o = 16; o > 0; o >>= 1) {
      s += __shfl_xor(s, o, 32);
      s2 += __shfl_xor(s2, o, 32);
    }
    if (lane == 0) { wsum[wave] = s; wsum2[wave] = s2; }
    __syncthreads();
    float mu = 0.f, m2 = 0.f;
#pragma unroll
    for (int w = 0; w < 6; ++w) { mu += wsum[w]; m2 += wsum2[w]; }
    mu *= (1.f / 192.f);
    m2 *= (1.f / 192.f);
    float var = m2 - mu * mu;
    acc += (v - mu) * rsqrtf(var + 1e-5f) * lng[c] + lnb[c];
    __syncthreads();
  }
  accs[c] = acc * (1.f / 128.f);
  __syncthreads();
  if (c < 3) {
    float s = bcls[c];
    for (int j = 0; j < 192; ++j) s += accs[j] * wcls[j * 3 + c];
    out[b * 3 + c] = s;
  }
}

// ---------------------------------------------------------------------------
// Packing / zero-fill helpers.
// ---------------------------------------------------------------------------
__global__ void cvt_bf16(const float* __restrict__ s, __bf16* __restrict__ d,
                         int n) {
  int i = blockIdx.x * 256 + threadIdx.x;
  if (i < n) d[i] = (__bf16)s[i];
}

__global__ void pack_conv(const float* __restrict__ w,
                          const float* __restrict__ bconv,
                          const float* __restrict__ g,
                          const float* __restrict__ bb,
                          const float* __restrict__ m,
                          const float* __restrict__ v, __bf16* __restrict__ wout,
                          float* __restrict__ bout, int Cout, int Cin, int KT) {
  int idx = blockIdx.x * 256 + threadIdx.x;
  int total = Cout * Cin * KT;
  if (idx < total) {
    int co = idx % Cout;
    int r = idx / Cout;
    int c = r % Cin;
    int t = r / Cin;
    float s = g ? g[co] * rsqrtf(v[co] + 1e-5f) : 1.f;
    wout[idx] = (__bf16)(w[(co * Cin + c) * KT + t] * s);
  }
  if (idx < Cout && bout) {
    float s = g ? g[idx] * rsqrtf(v[idx] + 1e-5f) : 1.f;
    float sh = g ? (bb[idx] - m[idx] * s) : 0.f;
    bout[idx] = bconv[idx] * s + sh;
  }
}

// zero 8 halo rows at each end of every batch: rows [0,8) and [LP-8,LP)
__global__ void zero_halo(__bf16* __restrict__ buf, int LP, int Cch, int nb) {
  int idx = blockIdx.x * 256 + threadIdx.x;
  int total = nb * 16 * Cch;
  if (idx >= total) return;
  int c = idx % Cch;
  int r2 = idx / Cch;
  int r = r2 % 16, b = r2 / 16;
  int row = (r < 8) ? r : (LP - 16 + r);
  buf[((size_t)b * LP + row) * Cch + c] = (__bf16)0.f;
}

// ---------------------------------------------------------------------------
extern "C" void kernel_launch(void* const* d_in, const int* in_sizes, int n_in,
                              void* d_out, int out_size, void* d_ws,
                              size_t ws_size, hipStream_t stream) {
  (void)in_sizes; (void)n_in; (void)out_size; (void)ws_size;
  const float* x     = (const float*)d_in[0];
  const float* w_adj = (const float*)d_in[1];
  const float* b_adj = (const float*)d_in[2];
  const float* w_c0 = (const float*)d_in[3];  const float* b_c0 = (const float*)d_in[4];
  const float* bn0g = (const float*)d_in[5];  const float* bn0b = (const float*)d_in[6];
  const float* bn0m = (const float*)d_in[7];  const float* bn0v = (const float*)d_in[8];
  const float* w_c1 = (const float*)d_in[9];  const float* b_c1 = (const float*)d_in[10];
  const float* bn1g = (const float*)d_in[11]; const float* bn1b = (const float*)d_in[12];
  const float* bn1m = (const float*)d_in[13]; const float* bn1v = (const float*)d_in[14];
  const float* w_c2 = (const float*)d_in[15]; const float* b_c2 = (const float*)d_in[16];
  const float* bn2g = (const float*)d_in[17]; const float* bn2b = (const float*)d_in[18];
  const float* bn2m = (const float*)d_in[19]; const float* bn2v = (const float*)d_in[20];
  const float* w_c3 = (const float*)d_in[21]; const float* b_c3 = (const float*)d_in[22];
  const float* bn3g = (const float*)d_in[23]; const float* bn3b = (const float*)d_in[24];
  const float* bn3m = (const float*)d_in[25]; const float* bn3v = (const float*)d_in[26];
  const float* w_pe = (const float*)d_in[27]; const float* b_pe = (const float*)d_in[28];
  const float* w_qk0 = (const float*)d_in[29]; const float* b_qk0 = (const float*)d_in[30];
  const float* w_qk1 = (const float*)d_in[31]; const float* b_qk1 = (const float*)d_in[32];
  const float* w_fc1 = (const float*)d_in[33]; const float* b_fc1 = (const float*)d_in[34];
  const float* w_fc2 = (const float*)d_in[35]; const float* b_fc2 = (const float*)d_in[36];
  const float* ln_g  = (const float*)d_in[37]; const float* ln_b = (const float*)d_in[38];
  const float* w_cls = (const float*)d_in[39]; const float* b_cls = (const float*)d_in[40];
  float* outp = (float*)d_out;

  char* ws = (char*)d_ws;
  size_t off = 0;
  auto alloc = [&](size_t bytes) -> void* {
    off = (off + 255) & ~(size_t)255;
    void* p = ws + off;
    off += bytes;
    return p;
  };
  const size_t B = 512, L0 = 2048, NT = 128, C = 192;
  const size_t LP = 1040;  // 8 + 1024 + 8 halo-padded rows
  __bf16* wadjb  = (__bf16*)alloc(1798ull * 2048 * 2);
  __bf16* adjout = (__bf16*)alloc(B * L0 * 2);
  __bf16* w0p = (__bf16*)alloc(15 * 64 * 2);    float* b0p = (float*)alloc(64 * 4);
  __bf16* w1p = (__bf16*)alloc(576 * 128 * 2);  float* b1p = (float*)alloc(128 * 4);
  __bf16* w2p = (__bf16*)alloc(640 * 128 * 2);  float* b2p = (float*)alloc(128 * 4);
  __bf16* w3p = (__bf16*)alloc(384 * 192 * 2);  float* b3p = (float*)alloc(192 * 4);
  __bf16* wpep = (__bf16*)alloc(2048ull * 192 * 2);
  __bf16* cat  = (__bf16*)alloc(B * LP * 256 * 2);  // halo-padded concat [x0|y3]
  __bf16* buf1 = (__bf16*)alloc(B * LP * 128 * 2);
  __bf16* buf2 = (__bf16*)alloc(B * LP * 128 * 2);
  __bf16* tokens = (__bf16*)alloc(B * NT * C * 2);
  __bf16* wqk0b = (__bf16*)alloc(192 * 384 * 2);
  __bf16* wqk1b = (__bf16*)alloc(192 * 384 * 2);
  __bf16* a1 = (__bf16*)alloc(B * NT * C * 2);
  __bf16* a2 = (__bf16*)alloc(B * NT * C * 2);
  __bf16* wfc1b = (__bf16*)alloc(192 * 264 * 2);
  __bf16* wfc2b = (__bf16*)alloc(264 * 192 * 2);
  __bf16* fc1out = (__bf16*)alloc(B * NT * 264 * 2);
  float* hbuf = (float*)alloc(B * NT * C * 4);
  (void)alloc(65536);  // slack: K-overhang async reads past last row land here

  auto cvt = [&](const float* s, __bf16* d, int n) {
    cvt_bf16<<<dim3((n + 255) / 256), dim3(256), 0, stream>>>(s, d, n);
  };
  cvt(w_adj, wadjb, 1798 * 2048);
  cvt(w_qk0, wqk0b, 192 * 384);
  cvt(w_qk1, wqk1b, 192 * 384);
  cvt(w_fc1, wfc1b, 192 * 264);
  cvt(w_fc2, wfc2b, 264 * 192);
  pack_conv<<<dim3((64 * 15 + 255) / 256), 256, 0, stream>>>(
      w_c0, b_c0, bn0g, bn0b, bn0m, bn0v, w0p, b0p, 64, 1, 15);
  pack_conv<<<dim3((128 * 576 + 255) / 256), 256, 0, stream>>>(
      w_c1, b_c1, bn1g, bn1b, bn1m, bn1v, w1p, b1p, 128, 64, 9);
  pack_conv<<<dim3((128 * 640 + 255) / 256), 256, 0, stream>>>(
      w_c2, b_c2, bn2g, bn2b, bn2m, bn2v, w2p, b2p, 128, 128, 5);
  pack_conv<<<dim3((192 * 384 + 255) / 256), 256, 0, stream>>>(
      w_c3, b_c3, bn3g, bn3b, bn3m, bn3v, w3p, b3p, 192, 128, 3);
  pack_conv<<<dim3((192 * 2048 + 255) / 256), 256, 0, stream>>>(
      w_pe, b_pe, nullptr, nullptr, nullptr, nullptr, wpep, nullptr, 192, 256, 8);
  // zero halo rows (read as conv padding)
  zero_halo<<<dim3((int)((B * 16 * 256 + 255) / 256)), 256, 0, stream>>>(
      cat, (int)LP, 256, (int)B);
  zero_halo<<<dim3((int)((B * 16 * 128 + 255) / 256)), 256, 0, stream>>>(
      buf1, (int)LP, 128, (int)B);
  zero_halo<<<dim3((int)((B * 16 * 128 + 255) / 256)), 256, 0, stream>>>(
      buf2, (int)LP, 128, (int)B);

  // adj projection: (512 x 1798) @ (1798 x 2048) -> bf16 (scalar staging, f32 A)
  conv_gemm<1, 1, float, __bf16, 0, false>
      <<<dim3(2048 / 64, 512 / 128), 256, 0, stream>>>(
          x, wadjb, b_adj, adjout, 1, 1, 0, 0, 0, 1798, 1798, 2048, 2048, 0);

  const int MC = (int)(B * 1024);  // 524288 conv output positions
  // c0: 1ch k15 s2 p7 (scalar staging: 2B-aligned gather) -> cat ch[0:64)
  conv_gemm<1, 15, __bf16, __bf16, 1, false><<<dim3(1, MC / 128), 256, 0, stream>>>(
      adjout, w0p, b0p, cat + 8 * 256, 1024, (int)LP, 2048, 2, 7, 1, 15, 64, 256, 0);
  // c1: 64ch k9 p4, async staging from halo-padded cat ch[0:64) -> buf1
  conv_gemm<64, 9, __bf16, __bf16, 1, true><<<dim3(2, MC / 128), 256, 0, stream>>>(
      cat, w1p, b1p, buf1 + 8 * 128, 1024, (int)LP, (int)LP, 1, 4, 256, 576, 128, 128, 0);
  // c2: 128ch k5 p2, async -> buf2
  conv_gemm<128, 5, __bf16, __bf16, 1, true><<<dim3(2, MC / 128), 256, 0, stream>>>(
      buf1, w2p, b2p, buf2 + 8 * 128, 1024, (int)LP, (int)LP, 1, 6, 128, 640, 128, 128, 0);
  // c3: 128ch k3 p1, async -> cat ch[64:256)
  conv_gemm<128, 3, __bf16, __bf16, 1, true><<<dim3(3, MC / 128), 256, 0, stream>>>(
      buf2, w3p, b3p, cat + 8 * 256, 1024, (int)LP, (int)LP, 1, 7, 128, 384, 192, 256, 64);
  const int MT = (int)(B * NT);  // 65536 tokens
  // pe: 256ch k8 s8, async -> bf16 tokens (B,128,192)
  conv_gemm<256, 8, __bf16, __bf16, 0, true><<<dim3(3, MT / 128), 256, 0, stream>>>(
      cat, wpep, b_pe, tokens, 128, 128, (int)LP, 8, 8, 256, 2048, 192, 192, 0);

  transformer_layer<<<dim3(512), 256, 0, stream>>>(tokens, wqk0b, b_qk0, a1);
  transformer_layer<<<dim3(512), 256, 0, stream>>>(a1, wqk1b, b_qk1, a2);

  // fc1 (gelu, async A) and fc2 (async A; K=264 overhang covered by slack)
  conv_gemm<1, 1, __bf16, __bf16, 2, true><<<dim3(5, MT / 128), 256, 0, stream>>>(
      tokens, wfc1b, b_fc1, fc1out, 1, 1, 0, 0, 0, 192, 192, 264, 264, 0);
  conv_gemm<1, 1, __bf16, float, 0, true><<<dim3(3, MT / 128), 256, 0, stream>>>(
      fc1out, wfc2b, b_fc2, hbuf, 1, 1, 0, 0, 0, 264, 264, 192, 192, 0);

  finalize_kernel<<<dim3(512), 192, 0, stream>>>(a2, hbuf, ln_g, ln_b, w_cls,
                                                 b_cls, outp);
}
